// RCBlock_14482629722250
// MI455X (gfx1250) — matrix operand order repaired and checked by reading, not compiled, not run on hardware
//
#include <hip/hip_runtime.h>

#define BB 8
#define CC 64
#define HH 128
#define WW 128
#define NPIX (BB*HH*WW)   // 131072

typedef __attribute__((ext_vector_type(16))) __bf16 v16bf;
typedef __attribute__((ext_vector_type(8)))  float  v8f;

// ---------------------------------------------------------------------------
// Kernel 1: fused 3x3 avg-pool (count_include_pad) + NCHW -> NHWC transpose of
// both p (pooled) and x (for the residual). NHWC makes the deformable-conv
// bilinear gathers (all 64 channels at one (y,x)) a contiguous 256B read.
// ---------------------------------------------------------------------------
__global__ void k_pool_nhwc(const float* __restrict__ x,
                            float* __restrict__ p_nhwc,
                            float* __restrict__ x_nhwc) {
    int idx = blockIdx.x * 256 + threadIdx.x;          // over B*C*H*W
    if (idx >= BB*CC*HH*WW) return;
    int w = idx & (WW-1);
    int h = (idx >> 7) & (HH-1);
    int c = (idx >> 14) & (CC-1);
    int b = idx >> 20;
    const float* xp = x + ((size_t)(b*CC + c)) * (HH*WW);
    float s = 0.f;
    #pragma unroll
    for (int dy = -1; dy <= 1; ++dy) {
        int yy = h + dy;
        if (yy < 0 || yy >= HH) continue;
        #pragma unroll
        for (int dx = -1; dx <= 1; ++dx) {
            int xx = w + dx;
            if (xx < 0 || xx >= WW) continue;
            s += xp[yy*WW + xx];
        }
    }
    int po = ((b*HH + h)*WW + w)*CC + c;
    p_nhwc[po] = s * (1.f/9.f);
    x_nhwc[po] = xp[h*WW + w];
}

// ---------------------------------------------------------------------------
// Kernel 2: pack weights into bf16 WMMA A-fragment order + fold BatchNorm.
// A-fragment layout (16-bit A, 16x32, ISA 7.12.2): lane%16 = M row,
// hi = lane>>4; element j<8 -> K = hi*8+j ; j>=8 -> K = 16 + hi*8 + (j-8).
// dwA: [mt=4][s=18][lane=32][j=16] over A = dw[64][576] (K = c*9 + tap).
// w1A/w2A: [mt=4][s=2][lane=32][j=16] over [64][64].
// ---------------------------------------------------------------------------
__global__ void k_prep(const float* __restrict__ dw, const float* __restrict__ w1,
                       const float* __restrict__ w2,
                       const float* __restrict__ gamma, const float* __restrict__ beta,
                       const float* __restrict__ rmean, const float* __restrict__ rvar,
                       __bf16* __restrict__ dwA, __bf16* __restrict__ w1A,
                       __bf16* __restrict__ w2A,
                       float* __restrict__ bnscale, float* __restrict__ bnbias) {
    int tid = threadIdx.x;   // single block of 256
    for (int i = tid; i < 4*18*32*16; i += 256) {
        int j = i & 15, lane = (i >> 4) & 31;
        int s = (i >> 9) % 18, mt = (i >> 9) / 18;
        int row = mt*16 + (lane & 15);
        int hi  = lane >> 4;
        int kr  = (j < 8) ? (hi*8 + j) : (16 + hi*8 + (j - 8));
        int k   = s*32 + kr;
        dwA[i] = (__bf16)dw[row*576 + k];
    }
    for (int i = tid; i < 4*2*32*16; i += 256) {
        int j = i & 15, lane = (i >> 4) & 31;
        int s = (i >> 9) & 1, mt = i >> 10;
        int row = mt*16 + (lane & 15);
        int hi  = lane >> 4;
        int kr  = (j < 8) ? (hi*8 + j) : (16 + hi*8 + (j - 8));
        int k   = s*32 + kr;
        w1A[i] = (__bf16)w1[row*64 + k];
        w2A[i] = (__bf16)w2[row*64 + k];
    }
    if (tid < 64) {
        float inv = gamma[tid] * rsqrtf(rvar[tid] + 1e-5f);
        bnscale[tid] = inv;
        bnbias[tid]  = beta[tid] - rmean[tid] * inv;
    }
}

// ---------------------------------------------------------------------------
// Kernel 3: offset-predicting 3x3 conv (18 out channels) on NHWC p.
// Weights staged in LDS (broadcast reads); one thread per pixel, 18 f32 accs.
// Output layout [pixel][18] so kernel 4 reads a pixel's offsets contiguously.
// ---------------------------------------------------------------------------
__global__ void k_offsets(const float* __restrict__ p_nhwc,
                          const float* __restrict__ off_w,
                          const float* __restrict__ off_b,
                          float* __restrict__ off_out) {
    __shared__ float lw[18*576];
    __shared__ float lb[18];
    for (int i = threadIdx.x; i < 18*576; i += 256) lw[i] = off_w[i];
    if (threadIdx.x < 18) lb[threadIdx.x] = off_b[threadIdx.x];
    __syncthreads();

    int pix = blockIdx.x * 256 + threadIdx.x;       // exact: NPIX/256 blocks
    int w = pix & (WW-1);
    int h = (pix >> 7) & (HH-1);
    int b = pix >> 14;

    float acc[18];
    #pragma unroll
    for (int o = 0; o < 18; ++o) acc[o] = lb[o];

    for (int t = 0; t < 9; ++t) {
        int yy = h + t/3 - 1, xx = w + t%3 - 1;
        if (yy < 0 || yy >= HH || xx < 0 || xx >= WW) continue;
        const float* base = p_nhwc + ((b*HH + yy)*WW + xx)*CC;
        for (int c = 0; c < CC; ++c) {
            float v = base[c];
            #pragma unroll
            for (int o = 0; o < 18; ++o)
                acc[o] += v * lw[o*576 + c*9 + t];   // LDS broadcast
        }
    }
    #pragma unroll
    for (int o = 0; o < 18; ++o) off_out[pix*18 + o] = acc[o];
}

// ---------------------------------------------------------------------------
// Kernel 4: fused deformable-conv GEMM + (w1 + BN + ReLU + residual) + w2.
// One block = 128 threads = 4 waves = 16 pixels of one image row.
// B-matrix layout (dense 16-bit B): lane%16 = N column, element j = K row (A's k order);
// Bs/B2/B3 are row-major [K][16] bf16 in LDS, gathered per element with stride 16.
// ---------------------------------------------------------------------------
__global__ void __launch_bounds__(128)
k_main(const float* __restrict__ p_nhwc, const float* __restrict__ x_nhwc,
       const float* __restrict__ off,
       const __bf16* __restrict__ dwA, const __bf16* __restrict__ w1A,
       const __bf16* __restrict__ w2A,
       const float* __restrict__ db, const float* __restrict__ b1,
       const float* __restrict__ bnscale, const float* __restrict__ bnbias,
       const float* __restrict__ b2, float* __restrict__ out) {
    __shared__ __bf16 Bs[576*16];   // sampled im2col tile, 18 KB
    __shared__ __bf16 B2[64*16];    // r  (deform-conv result) as bf16
    __shared__ __bf16 B3[64*16];    // x + relu(bn(w1 r)) as bf16

    int tid  = threadIdx.x;
    int wv   = tid >> 5, lane = tid & 31;
    int bid  = blockIdx.x;                  // B*H*(W/16) = 8192 blocks
    int xt   = bid & 7;
    int y    = (bid >> 3) & (HH-1);
    int b    = bid >> 10;
    int x0   = xt * 16;

    // ---- build sampled B (576 x 16): bilinear gather from NHWC p ----
    for (int pp = wv; pp < 144; pp += 4) {          // (pixel n, tap) pairs
        int n = pp / 9, tap = pp % 9;
        int px  = x0 + n;
        int pix = (b*HH + y)*WW + px;
        float oy = off[pix*18 + tap*2 + 0];
        float ox = off[pix*18 + tap*2 + 1];
        float sy = (float)y  + (float)(tap/3 - 1) + oy;
        float sx = (float)px + (float)(tap%3 - 1) + ox;
        float fy = floorf(sy), fx = floorf(sx);
        float wy = sy - fy,  wx = sx - fx;
        int   iy = (int)fy,  ix = (int)fx;
        int   c0 = lane * 2;                 // each lane covers 2 channels
        float v0 = 0.f, v1 = 0.f;
        #pragma unroll
        for (int corner = 0; corner < 4; ++corner) {
            int   cy  = iy + (corner >> 1), cx = ix + (corner & 1);
            float wgt = ((corner >> 1) ? wy : 1.f - wy) *
                        ((corner & 1)  ? wx : 1.f - wx);
            if (cy >= 0 && cy < HH && cx >= 0 && cx < WW) {   // wave-uniform
                const float2 pv =
                    *(const float2*)(p_nhwc + ((b*HH + cy)*WW + cx)*CC + c0);
                v0 += wgt * pv.x;
                v1 += wgt * pv.y;
            }
        }
        Bs[( c0     *9 + tap)*16 + n] = (__bf16)v0;   // K = c*9 + tap
        Bs[((c0 + 1)*9 + tap)*16 + n] = (__bf16)v1;
    }
    __syncthreads();

    // ---- GEMM1: r(64x16) = dw(64x576) x Bs(576x16), f32 accumulate ----
    v8f acc = {};
    for (int s = 0; s < 18; ++s) {
        v16bf a  = *(const v16bf*)(dwA + (size_t)((wv*18 + s)*32 + lane)*16);
        v16bf bm;
        #pragma unroll
        for (int j = 0; j < 16; ++j)
            bm[j] = Bs[(s*32 + ((j < 8) ? ((lane >> 4)*8 + j) : (16 + (lane >> 4)*8 + (j - 8))))*16 + (lane & 15)];
        acc = __builtin_amdgcn_wmma_f32_16x16x32_bf16(false, a, false, bm,
                                                      (short)0, acc, false, false);
    }
    int hi = lane >> 4, n = lane & 15;
    #pragma unroll
    for (int v = 0; v < 8; ++v) {                 // C layout: M = v + 8*hi
        int row = wv*16 + hi*8 + v;
        B2[row*16 + n] = (__bf16)(acc[v] + db[row]);
    }
    __syncthreads();

    // ---- GEMM2: h = w1 x r ; BN(eval) + ReLU + residual(x) ----
    v8f acc2 = {};
    for (int s = 0; s < 2; ++s) {
        v16bf a  = *(const v16bf*)(w1A + (size_t)((wv*2 + s)*32 + lane)*16);
        v16bf bm;
        #pragma unroll
        for (int j = 0; j < 16; ++j)
            bm[j] = B2[(s*32 + ((j < 8) ? (hi*8 + j) : (16 + hi*8 + (j - 8))))*16 + n];
        acc2 = __builtin_amdgcn_wmma_f32_16x16x32_bf16(false, a, false, bm,
                                                       (short)0, acc2, false, false);
    }
    int pixn = (b*HH + y)*WW + x0 + n;
    #pragma unroll
    for (int v = 0; v < 8; ++v) {
        int row = wv*16 + hi*8 + v;
        float h = acc2[v] + b1[row];
        h = h * bnscale[row] + bnbias[row];
        h = fmaxf(h, 0.f);
        float t = x_nhwc[pixn*CC + row] + h;      // contiguous per lane
        B3[row*16 + n] = (__bf16)t;
    }
    __syncthreads();

    // ---- GEMM3: out = w2 x t + b2, store NCHW f32 ----
    v8f acc3 = {};
    for (int s = 0; s < 2; ++s) {
        v16bf a  = *(const v16bf*)(w2A + (size_t)((wv*2 + s)*32 + lane)*16);
        v16bf bm;
        #pragma unroll
        for (int j = 0; j < 16; ++j)
            bm[j] = B3[(s*32 + ((j < 8) ? (hi*8 + j) : (16 + hi*8 + (j - 8))))*16 + n];
        acc3 = __builtin_amdgcn_wmma_f32_16x16x32_bf16(false, a, false, bm,
                                                       (short)0, acc3, false, false);
    }
    #pragma unroll
    for (int v = 0; v < 8; ++v) {
        int row = wv*16 + hi*8 + v;
        out[((size_t)(b*CC + row)*HH + y)*WW + x0 + n] = acc3[v] + b2[row];
    }
}

// ---------------------------------------------------------------------------
extern "C" void kernel_launch(void* const* d_in, const int* in_sizes, int n_in,
                              void* d_out, int out_size, void* d_ws, size_t ws_size,
                              hipStream_t stream) {
    (void)in_sizes; (void)n_in; (void)out_size; (void)ws_size;
    const float* x     = (const float*)d_in[0];
    const float* off_w = (const float*)d_in[1];
    const float* off_b = (const float*)d_in[2];
    const float* dw    = (const float*)d_in[3];
    const float* db    = (const float*)d_in[4];
    const float* w1    = (const float*)d_in[5];
    const float* b1    = (const float*)d_in[6];
    const float* gamma = (const float*)d_in[7];
    const float* beta  = (const float*)d_in[8];
    const float* rmean = (const float*)d_in[9];
    const float* rvar  = (const float*)d_in[10];
    const float* w2    = (const float*)d_in[11];
    const float* b2    = (const float*)d_in[12];
    float* out = (float*)d_out;

    char* ws = (char*)d_ws;
    float*  p_nhwc = (float*)(ws);                              // 33,554,432 B
    float*  x_nhwc = (float*)(ws + 33554432);                   // 33,554,432 B
    float*  offbuf = (float*)(ws + 67108864);                   //  9,437,184 B
    __bf16* dwA    = (__bf16*)(ws + 76546048);                  //     73,728 B
    __bf16* w1A    = (__bf16*)(ws + 76546048 + 73728);          //      8,192 B
    __bf16* w2A    = (__bf16*)(ws + 76546048 + 73728 + 8192);   //      8,192 B
    float*  bnscale = (float*)(ws + 76546048 + 73728 + 16384);  //        256 B
    float*  bnbias  = bnscale + 64;                             //        256 B

    k_pool_nhwc<<<(BB*CC*HH*WW)/256, 256, 0, stream>>>(x, p_nhwc, x_nhwc);
    k_prep<<<1, 256, 0, stream>>>(dw, w1, w2, gamma, beta, rmean, rvar,
                                  dwA, w1A, w2A, bnscale, bnbias);
    k_offsets<<<NPIX/256, 256, 0, stream>>>(p_nhwc, off_w, off_b, offbuf);
    k_main<<<BB*HH*(WW/16), 128, 0, stream>>>(p_nhwc, x_nhwc, offbuf,
                                              dwA, w1A, w2A, db, b1,
                                              bnscale, bnbias, b2, out);
}
